// HmmLm_53102975648281
// MI455X (gfx1250) — compile-verified, matrix operand order checked by reading, not run on hardware
//
#include <hip/hip_runtime.h>
#include <math.h>

typedef __attribute__((ext_vector_type(16))) _Float16 v16h;
typedef __attribute__((ext_vector_type(8)))  _Float16 v8h;
typedef __attribute__((ext_vector_type(4)))  _Float16 v4h;
typedef __attribute__((ext_vector_type(8)))  float    v8f;
typedef __attribute__((ext_vector_type(4)))  float    v4f;
typedef unsigned int u32x4 __attribute__((ext_vector_type(4)));
typedef int          i32x8 __attribute__((ext_vector_type(8)));
typedef int          i32x4 __attribute__((ext_vector_type(4)));

#define Cdim 1024
#define Hdim 256
#define Vdim 10000
#define Ndim 16
#define Tdim 256

__device__ __forceinline__ v16h cat8(v8h lo, v8h hi) {
  v16h r;
#pragma unroll
  for (int i = 0; i < 8; ++i) { r[i] = lo[i]; r[i + 8] = hi[i]; }
  return r;
}

// ---------------------------------------------------------------------------
// LDS-staged WMMA GEMM: Cout = epilogue(A[f32 MxK] @ B[f32 KxN] + bias)
// 256 threads, 64x64 block tile, 8 waves each computing one 16x32 strip
// (two 16x16 WMMA accumulators). K multiple of 32, M multiple of 64.
// flags: bit0 = relu, bit1 = add residual.
// ---------------------------------------------------------------------------
#define BM 64
#define BN 64
#define BK 32

__global__ __launch_bounds__(256) void gemm_wmma(
    const float* __restrict__ A, const float* __restrict__ B,
    const float* __restrict__ bias, const float* __restrict__ resid,
    float* __restrict__ Cout, int M, int N, int K, int flags) {
  __shared__ _Float16 As[BM * BK];   // [row][k]
  __shared__ _Float16 Bs[BN * BK];   // [col][k]  (transposed)

  const int tid  = threadIdx.x;
  const int lane = tid & 31, wave = tid >> 5;
  const int wm = wave & 3, wn = wave >> 2;
  const int l15 = lane & 15;
  const int khalf = (lane & 16) ? 8 : 0;
  const int hi8   = (lane & 16) ? 8 : 0;
  const int m0 = blockIdx.y * BM;
  const int n0 = blockIdx.x * BN;
  const bool fullN = (n0 + BN) <= N;   // block-uniform

  v8f acc0 = {}, acc1 = {};

  for (int k0 = 0; k0 < K; k0 += BK) {
    // ---- stage A tile: 64x32, coalesced b128 loads, f32 -> f16
#pragma unroll
    for (int i = 0; i < 2; ++i) {
      const int c  = tid + i * 256;       // 0..511 float4 chunks
      const int r  = c >> 3;
      const int k4 = (c & 7) << 2;
      v4f v = *(const v4f*)(A + (size_t)(m0 + r) * K + k0 + k4);
      v4h h; h[0] = (_Float16)v[0]; h[1] = (_Float16)v[1];
             h[2] = (_Float16)v[2]; h[3] = (_Float16)v[3];
      *(v4h*)(As + r * BK + k4) = h;
    }
    // ---- stage B tile: 32x64 -> Bs[col][k]
    if (fullN) {
#pragma unroll
      for (int i = 0; i < 2; ++i) {
        const int c   = tid + i * 256;    // 0..511
        const int k   = c >> 4;
        const int nn4 = (c & 15) << 2;
        v4f v = *(const v4f*)(B + (size_t)(k0 + k) * N + n0 + nn4);
        Bs[(nn4 + 0) * BK + k] = (_Float16)v[0];
        Bs[(nn4 + 1) * BK + k] = (_Float16)v[1];
        Bs[(nn4 + 2) * BK + k] = (_Float16)v[2];
        Bs[(nn4 + 3) * BK + k] = (_Float16)v[3];
      }
    } else {                              // only last N-tile (V=10000 tail)
#pragma unroll
      for (int i = 0; i < 8; ++i) {
        const int c  = tid + i * 256;     // 0..2047
        const int k  = c >> 6;
        const int nn = c & 63;
        const int col  = n0 + nn;
        const int colc = col < N ? col : (N - 1);   // branchless clamp
        float v = B[(size_t)(k0 + k) * N + colc];
        Bs[nn * BK + k] = (_Float16)(col < N ? v : 0.f);
      }
    }
    __syncthreads();

    // ---- fragments + 2 WMMAs per wave
    const _Float16* Ap  = As + (wm * 16 + l15) * BK + khalf;
    const _Float16* Bp0 = Bs + (wn * 32 + l15) * BK + khalf;
    const _Float16* Bp1 = Bs + (wn * 32 + 16 + l15) * BK + khalf;
    v16h av  = cat8(*(const v8h*)Ap,  *(const v8h*)(Ap + 16));
    v16h bv0 = cat8(*(const v8h*)Bp0, *(const v8h*)(Bp0 + 16));
    v16h bv1 = cat8(*(const v8h*)Bp1, *(const v8h*)(Bp1 + 16));
    acc0 = __builtin_amdgcn_wmma_f32_16x16x32_f16(false, av, false, bv0,
                                                  (short)0, acc0, false, false);
    acc1 = __builtin_amdgcn_wmma_f32_16x16x32_f16(false, av, false, bv1,
                                                  (short)0, acc1, false, false);
    __syncthreads();
  }

  // ---- epilogue
#pragma unroll
  for (int half = 0; half < 2; ++half) {
    const v8f& acc = half ? acc1 : acc0;
    const int col = n0 + wn * 32 + half * 16 + l15;
    if (col < N) {
      const float bb = bias ? bias[col] : 0.f;
#pragma unroll
      for (int v = 0; v < 8; ++v) {
        const int mrow = m0 + wm * 16 + v + hi8;
        float x = acc[v] + bb;
        if (flags & 1) x = fmaxf(x, 0.f);
        if (flags & 2) x += resid[(size_t)mrow * N + col];
        Cout[(size_t)mrow * N + col] = x;
      }
    }
  }
}

// matvec: out[r] = dot(A[r,:], w) + b[0]; one wave per row
__global__ __launch_bounds__(256) void matvec(const float* __restrict__ A,
                                              const float* __restrict__ w,
                                              const float* __restrict__ b,
                                              float* __restrict__ out, int K) {
  const int row = blockIdx.x * 8 + (threadIdx.x >> 5);
  const int lane = threadIdx.x & 31;
  float s = 0.f;
  for (int k = lane; k < K; k += 32) s += A[(size_t)row * K + k] * w[k];
#pragma unroll
  for (int off = 16; off; off >>= 1) s += __shfl_xor(s, off);
  if (lane == 0) out[row] = s + b[0];
}

// per-row logsumexp of X [R x Ncols], one block per row
__global__ __launch_bounds__(256) void row_lse(const float* __restrict__ X,
                                               float* __restrict__ lse, int Ncols) {
  const int r = blockIdx.x, tid = threadIdx.x;
  __shared__ float sm[256];
  const float* row = X + (size_t)r * Ncols;
  float m = -INFINITY;
  for (int j = tid; j < Ncols; j += 256) m = fmaxf(m, row[j]);
  sm[tid] = m; __syncthreads();
  for (int s = 128; s; s >>= 1) { if (tid < s) sm[tid] = fmaxf(sm[tid], sm[tid + s]); __syncthreads(); }
  m = sm[0]; __syncthreads();
  float sum = 0.f;
  for (int j = tid; j < Ncols; j += 256) sum += expf(row[j] - m);
  sm[tid] = sum; __syncthreads();
  for (int s = 128; s; s >>= 1) { if (tid < s) sm[tid] += sm[tid + s]; __syncthreads(); }
  if (tid == 0) lse[r] = m + logf(sm[0]);
}

// t_max[j] = max_i (logitsT[i,j] - lseT[i]); one block per column j
__global__ __launch_bounds__(256) void col_max(const float* __restrict__ X,
                                               const float* __restrict__ rowlse,
                                               float* __restrict__ cmax) {
  const int j = blockIdx.x, tid = threadIdx.x;
  __shared__ float sm[256];
  float m = -INFINITY;
  for (int i = tid; i < Cdim; i += 256) m = fmaxf(m, X[(size_t)i * Cdim + j] - rowlse[i]);
  sm[tid] = m; __syncthreads();
  for (int s = 128; s; s >>= 1) { if (tid < s) sm[tid] = fmaxf(sm[tid], sm[tid + s]); __syncthreads(); }
  if (tid == 0) cmax[j] = sm[0];
}

// Qh[j*1024+i] = f16( exp(logitsT[i,j] - lseT[i] - tmax[j]) )
__global__ __launch_bounds__(256) void make_q(const float* __restrict__ X,
                                              const float* __restrict__ rowlse,
                                              const float* __restrict__ cmax,
                                              _Float16* __restrict__ Q) {
  const int idx = blockIdx.x * 256 + threadIdx.x;
  const int j = idx >> 10, i = idx & 1023;
  Q[idx] = (_Float16)expf(X[(size_t)i * Cdim + j] - rowlse[i] - cmax[j]);
}

// obs[(n*T+t)*C + c] = E[c, text[n,t]] - lseE[c]; one block per (n,t)
__global__ __launch_bounds__(256) void make_obs(const float* __restrict__ E,
                                                const float* __restrict__ lseE,
                                                const int* __restrict__ text,
                                                float* __restrict__ obs) {
  const int nt = blockIdx.x;
  int v = text[nt];
  if (v < 0) v = 0; if (v >= Vdim) v = Vdim - 1;
  for (int c = threadIdx.x; c < Cdim; c += 256)
    obs[(size_t)nt * Cdim + c] = E[(size_t)c * Vdim + v] - lseE[c];
}

// alpha0[n,c] = (slogit[c] - lseS) + obs[n,0,c]
__global__ __launch_bounds__(256) void init_alpha(const float* __restrict__ slogit,
                                                  const float* __restrict__ lseS,
                                                  const float* __restrict__ obs,
                                                  float* __restrict__ alpha0) {
  const int idx = blockIdx.x * 256 + threadIdx.x;
  const int n = idx >> 10, c = idx & 1023;
  alpha0[idx] = (slogit[c] - lseS[0]) + obs[(size_t)(n * Tdim) * Cdim + c];
}

__global__ void zero_counter(unsigned int* c) { *c = 0u; }

// ---------------------------------------------------------------------------
// Persistent forward recursion: 8 workgroups x 256 threads (8 waves).
// Block b owns j in [b*128, b*128+128); its 128x1024 f16 Q slice (256 KB of
// the 320 KB WGP LDS) is loaded once via the Tensor Data Mover, then each
// step rebuilds the 32 KB P panel and runs one 16x16 j-tile per wave
// (32 chained WMMAs, K=1024), with a device-wide atomic barrier per step.
// ---------------------------------------------------------------------------
__global__ __launch_bounds__(256, 1) void forward_persistent(
    const _Float16* __restrict__ Qg, const float* __restrict__ tmax,
    const float* __restrict__ obs, float* __restrict__ alphaA,
    float* __restrict__ alphaB, unsigned int* __restrict__ counter) {
  extern __shared__ char smem[];
  _Float16* Qs  = (_Float16*)smem;                                 // 128*1024 halves
  _Float16* Ps  = (_Float16*)(smem + 128 * 1024 * 2);              // 16*1024 halves
  float*    red = (float*)(smem + 128 * 1024 * 2 + 16 * 1024 * 2); // 256 f32
  float*    am  = red + 256;                                       // 16 f32

  const int tid = threadIdx.x;
  const int lane = tid & 31, wave = tid >> 5;
  const int j0 = blockIdx.x * 128;
  const int l15 = lane & 15;
  const int khalf = (lane & 16) ? 8 : 0;

  // ---- preload Q slice once: TDM if available, else vector copy
#if defined(__has_builtin) && __has_builtin(__builtin_amdgcn_tensor_load_to_lds) && \
    __has_builtin(__builtin_amdgcn_s_wait_tensorcnt)
  if (wave == 0) {
    const unsigned long long gaddr =
        (unsigned long long)(const void*)(Qg + (size_t)j0 * Cdim);
    const unsigned ldsaddr = (unsigned)(unsigned long long)(void*)Qs; // low 32b = LDS offset
    u32x4 g0 = {};
    g0[0] = 1u;                                        // count=1, user descriptor
    g0[1] = ldsaddr;                                   // lds_addr
    g0[2] = (unsigned)(gaddr & 0xFFFFFFFFu);           // global_addr[31:0]
    g0[3] = (unsigned)((gaddr >> 32) & 0x01FFFFFFu)    // global_addr[56:32]
            | (2u << 30);                              // type = 2 ("image")
    i32x8 g1 = {};
    g1[0] = 0x00010000;          // workgroup_mask=0, data_size=1 (2 bytes)
    g1[1] = (int)(1024u << 16);  // tensor_dim0 = 1024 (low 16 @ bits 63:48)
    g1[2] = (int)(1024u << 16);  // tensor_dim1 = 1024 (low 16 @ bits 95:80)
    g1[3] = (int)(1024u << 16);  // tile_dim0  = 1024 (bits 127:112)
    g1[4] = 128;                 // tile_dim1  = 128, tile_dim2 = 0
    g1[5] = 1024;                // tensor_dim0_stride = 1024 (low 32)
    g1[6] = 0; g1[7] = 0;
    i32x4 gz = {};
#if __has_include(<hip/amd_detail/amd_gfx1250_TDM.h>)
    i32x8 gz8 = {};
    __builtin_amdgcn_tensor_load_to_lds(g0, g1, gz, gz, gz8, 0);   // clang-23 form
#else
    __builtin_amdgcn_tensor_load_to_lds(g0, g1, gz, gz, 0);        // ROCm 7.2 form
#endif
    __builtin_amdgcn_s_wait_tensorcnt(0);
  }
#else
  {
    const v8h* src = (const v8h*)(Qg + (size_t)j0 * Cdim);
    v8h* dst = (v8h*)Qs;
    for (int p = tid; p < 128 * Cdim / 8; p += 256) dst[p] = src[p];
  }
#endif
  __syncthreads();

  float* cur = alphaA;
  float* nxt = alphaB;
  const int n_ = tid & 15, chunk = tid >> 4;   // 16 chunks of 64 per batch row

  for (int step = 1; step < Tdim; ++step) {
    // ---- a_max[n] over 1024 states (vectorized)
    const float* arow = cur + n_ * Cdim + chunk * 64;
    float m = -INFINITY;
#pragma unroll
    for (int q = 0; q < 64; q += 4) {
      v4f v = *(const v4f*)(arow + q);
      m = fmaxf(m, fmaxf(fmaxf(v[0], v[1]), fmaxf(v[2], v[3])));
    }
    red[tid] = m;
    __syncthreads();
    if (tid < 16) {
      float mm = red[tid];
      for (int c2 = 1; c2 < 16; ++c2) mm = fmaxf(mm, red[(c2 << 4) | tid]);
      am[tid] = mm;
    }
    __syncthreads();
    // ---- P panel: exp(alpha - a_max) -> f16 (b128 load / b64 store)
    {
      const float mm = am[n_];
      _Float16* prow = Ps + n_ * Cdim + chunk * 64;
#pragma unroll
      for (int q = 0; q < 64; q += 4) {
        v4f v = *(const v4f*)(arow + q);
        v4h h;
        h[0] = (_Float16)__expf(v[0] - mm); h[1] = (_Float16)__expf(v[1] - mm);
        h[2] = (_Float16)__expf(v[2] - mm); h[3] = (_Float16)__expf(v[3] - mm);
        *(v4h*)(prow + q) = h;
      }
    }
    __syncthreads();

    // ---- GEMM: this wave's 16x16 tile, K = 1024
    v8f acc = {};
    const _Float16* Arow = Ps + l15 * Cdim;
    const _Float16* Brow = Qs + (wave * 16 + l15) * Cdim;
    for (int kb = 0; kb < Cdim; kb += 32) {
      const int ko = kb + khalf;
      v16h av = cat8(*(const v8h*)(Arow + ko), *(const v8h*)(Arow + ko + 16));
      v16h bv = cat8(*(const v8h*)(Brow + ko), *(const v8h*)(Brow + ko + 16));
      acc = __builtin_amdgcn_wmma_f32_16x16x32_f16(false, av, false, bv,
                                                   (short)0, acc, false, false);
    }

    // ---- epilogue: back to log space + t_max + obs
    const int j = j0 + wave * 16 + l15;
    const float tmj = tmax[j];
#pragma unroll
    for (int v = 0; v < 8; ++v) {
      const int nn = v + ((lane & 16) ? 8 : 0);
      float x = logf(fmaxf(acc[v], 1e-37f)) + am[nn] + tmj +
                obs[(size_t)(nn * Tdim + step) * Cdim + j];
      nxt[nn * Cdim + j] = x;
    }

    // ---- device-wide step barrier (8 resident workgroups)
    __threadfence();
    __syncthreads();
    if (tid == 0) {
      atomicAdd(counter, 1u);
      const unsigned int target = 8u * (unsigned int)step;
      while (__hip_atomic_load(counter, __ATOMIC_ACQUIRE,
                               __HIP_MEMORY_SCOPE_AGENT) < target) { }
    }
    __syncthreads();

    float* t2 = cur; cur = nxt; nxt = t2;
  }
}

// final: sum_n logsumexp_j alpha[n,j]
__global__ __launch_bounds__(256) void final_reduce(const float* __restrict__ alpha,
                                                    float* __restrict__ out) {
  __shared__ float sm[256];
  __shared__ float tot;
  const int tid = threadIdx.x;
  if (tid == 0) tot = 0.f;
  __syncthreads();
  for (int n = 0; n < Ndim; ++n) {
    const float* a = alpha + (size_t)n * Cdim;
    float m = -INFINITY;
    for (int j = tid; j < Cdim; j += 256) m = fmaxf(m, a[j]);
    sm[tid] = m; __syncthreads();
    for (int s = 128; s; s >>= 1) { if (tid < s) sm[tid] = fmaxf(sm[tid], sm[tid + s]); __syncthreads(); }
    m = sm[0]; __syncthreads();
    float sum = 0.f;
    for (int j = tid; j < Cdim; j += 256) sum += expf(a[j] - m);
    sm[tid] = sum; __syncthreads();
    for (int s = 128; s; s >>= 1) { if (tid < s) sm[tid] += sm[tid + s]; __syncthreads(); }
    if (tid == 0) tot += m + logf(sm[0]);
    __syncthreads();
  }
  if (tid == 0) out[0] = tot;
}

// ---------------------------------------------------------------------------
extern "C" void kernel_launch(void* const* d_in, const int* in_sizes, int n_in,
                              void* d_out, int out_size, void* d_ws, size_t ws_size,
                              hipStream_t stream) {
  const int*   text      = (const int*)d_in[0];
  const float* start_emb = (const float*)d_in[1];
  const float* sw1 = (const float*)d_in[2];  const float* sb1 = (const float*)d_in[3];
  const float* sw2 = (const float*)d_in[4];  const float* sb2 = (const float*)d_in[5];
  const float* sw3 = (const float*)d_in[6];  const float* sb3 = (const float*)d_in[7];
  const float* state_emb = (const float*)d_in[8];
  const float* tw1 = (const float*)d_in[9];  const float* tb1 = (const float*)d_in[10];
  const float* tw2 = (const float*)d_in[11]; const float* tb2 = (const float*)d_in[12];
  const float* tw3 = (const float*)d_in[13]; const float* tb3 = (const float*)d_in[14];
  const float* pre_emb = (const float*)d_in[15];
  const float* ew1 = (const float*)d_in[16]; const float* eb1 = (const float*)d_in[17];
  const float* ew2 = (const float*)d_in[18]; const float* eb2 = (const float*)d_in[19];
  const float* ew3 = (const float*)d_in[20]; const float* eb3 = (const float*)d_in[21];

  char* ws = (char*)d_ws;
  auto alloc = [&](size_t bytes) -> char* {
    char* p = ws;
    ws += (bytes + 255) & ~(size_t)255;
    return p;
  };
  float*     E       = (float*)alloc((size_t)Cdim * Vdim * 4);
  float*     obs     = (float*)alloc((size_t)Ndim * Tdim * Cdim * 4);
  float*     logitsT = (float*)alloc((size_t)Cdim * Cdim * 4);
  _Float16*  Qh      = (_Float16*)alloc((size_t)Cdim * Cdim * 2);
  float*     hbuf    = (float*)alloc((size_t)Cdim * Hdim * 4);
  float*     rbuf    = (float*)alloc((size_t)Cdim * Hdim * 4);
  float*     slogit  = (float*)alloc(Cdim * 4);
  float*     lseS    = (float*)alloc(256);
  float*     lseT    = (float*)alloc(Cdim * 4);
  float*     lseE    = (float*)alloc(Cdim * 4);
  float*     tmax    = (float*)alloc(Cdim * 4);
  float*     alphaA  = (float*)alloc((size_t)Ndim * Cdim * 4);
  float*     alphaB  = (float*)alloc((size_t)Ndim * Cdim * 4);
  unsigned*  counter = (unsigned*)alloc(256);

  const dim3 b256(256);
  const dim3 gHH(Hdim / BN, Cdim / BM);   // [1024,256] outputs

  // start head
  gemm_wmma<<<gHH, b256, 0, stream>>>(start_emb, sw1, sb1, nullptr, hbuf, Cdim, Hdim, Hdim, 1);
  gemm_wmma<<<gHH, b256, 0, stream>>>(hbuf, sw2, sb2, start_emb, rbuf, Cdim, Hdim, Hdim, 3);
  matvec<<<Cdim / 8, b256, 0, stream>>>(rbuf, sw3, sb3, slogit, Hdim);
  row_lse<<<1, b256, 0, stream>>>(slogit, lseS, Cdim);

  // transition head
  gemm_wmma<<<gHH, b256, 0, stream>>>(state_emb, tw1, tb1, nullptr, hbuf, Cdim, Hdim, Hdim, 1);
  gemm_wmma<<<gHH, b256, 0, stream>>>(hbuf, tw2, tb2, state_emb, rbuf, Cdim, Hdim, Hdim, 3);
  gemm_wmma<<<dim3(Cdim / BN, Cdim / BM), b256, 0, stream>>>(rbuf, tw3, tb3, nullptr, logitsT, Cdim, Cdim, Hdim, 0);
  row_lse<<<Cdim, b256, 0, stream>>>(logitsT, lseT, Cdim);
  col_max<<<Cdim, b256, 0, stream>>>(logitsT, lseT, tmax);
  make_q<<<(Cdim * Cdim) / 256, b256, 0, stream>>>(logitsT, lseT, tmax, Qh);

  // emission head
  gemm_wmma<<<gHH, b256, 0, stream>>>(pre_emb, ew1, eb1, nullptr, hbuf, Cdim, Hdim, Hdim, 1);
  gemm_wmma<<<gHH, b256, 0, stream>>>(hbuf, ew2, eb2, pre_emb, rbuf, Cdim, Hdim, Hdim, 3);
  gemm_wmma<<<dim3((Vdim + BN - 1) / BN, Cdim / BM), b256, 0, stream>>>(rbuf, ew3, eb3, nullptr, E, Cdim, Vdim, Hdim, 0);
  row_lse<<<Cdim, b256, 0, stream>>>(E, lseE, Vdim);
  make_obs<<<Ndim * Tdim, b256, 0, stream>>>(E, lseE, text, obs);

  // recursion
  init_alpha<<<(Ndim * Cdim) / 256, b256, 0, stream>>>(slogit, lseS, obs, alphaA);
  zero_counter<<<1, 1, 0, stream>>>(counter);
  const size_t shmem = 128 * 1024 * 2 + 16 * 1024 * 2 + 256 * 4 + 16 * 4; // ~295 KB of 320 KB WGP LDS
  forward_persistent<<<8, b256, shmem, stream>>>(Qh, tmax, obs, alphaA, alphaB, counter);

  // T-1 = 255 steps -> final alpha lives in alphaB
  final_reduce<<<1, b256, 0, stream>>>(alphaB, (float*)d_out);
}
